// iMeshSegNet_37254546325701
// MI455X (gfx1250) — matrix operand order repaired and checked by reading, not compiled, not run on hardware
//
#include <hip/hip_runtime.h>
#include <hip/hip_bf16.h>
#include <math.h>

#define EPS_BN 1e-5f

typedef _Float16 f16;
typedef __attribute__((ext_vector_type(16))) _Float16 v16h;
typedef __attribute__((ext_vector_type(4)))  _Float16 hx4;
typedef __attribute__((ext_vector_type(8)))  float    v8f;
typedef __attribute__((ext_vector_type(4)))  float    fx4;
typedef __attribute__((ext_vector_type(8)))  unsigned v8u;

__device__ __forceinline__ float apply_act(float y, int act) {
  if (act == 1) return fmaxf(y, 0.f);
  if (act == 2) return (y < 0.f) ? 0.2f * y : y;
  return y;
}

#define BM 32
#define BN 128
#define BK 32
#define APAD 4   // As row stride 36 f16 = 72B  (8B aligned)
#define BPADF 4  // Bs row stride 132 f32 = 528B (16B aligned; 2addr offset 132 dw < 255)

// ---------------------------------------------------------------------------
// WMMA GEMM: Y[b] = act(bn(A[b](MxK) * X[b](K x Ncols) + convb))
// 32x128 tile, 8 waves (2M x 4N), 2 WMMA per wave per 32-K step.
// B staged in LDS as raw f32 (b128 copy), converted to f16 at fragment build
// via v_cvt_pk_f16_f32. Double-buffered LDS, register prefetch, branch-free
// clamped loads.
// ---------------------------------------------------------------------------
__global__ __launch_bounds__(256) void gemm_bn_act(
    const float* __restrict__ A, long long strideA,
    const float* __restrict__ X, long long strideX,
    float* __restrict__ Y, long long strideY,
    const float* __restrict__ convb,
    const float* __restrict__ bng, const float* __restrict__ bnb,
    const float* __restrict__ bnm, const float* __restrict__ bnv,
    int M, int K, int Ncols, int act)
{
  __shared__ f16   As[2][BM][BK + APAD];
  __shared__ float Bs[2][BK][BN + BPADF];

  const int b     = blockIdx.z;
  const int tileN = blockIdx.x * BN;
  const int tileM = blockIdx.y * BM;
  const int tid   = threadIdx.x;
  const int lane  = tid & 31;
  const int wave  = tid >> 5;
  const int wm    = wave & 1;
  const int wn    = wave >> 1;
  const int half  = lane >> 4;
  const int mr    = lane & 15;

  const float* Ab = A + (size_t)b * strideA;
  const float* Xb = X + (size_t)b * strideX;
  float*       Yb = Y + (size_t)b * strideY;

  const int arow  = tid >> 3;          // 0..31
  const int acol4 = (tid & 7) << 2;    // 0,4,..,28
  const int agr   = tileM + arow;

  float ra[4];
  fx4   rb[4];
  const fx4 zero4 = {0.f, 0.f, 0.f, 0.f};

  auto loadA = [&](int k0) {
#pragma unroll
    for (int j = 0; j < 4; ++j) {
      int gc = k0 + acol4 + j;
      bool ok = (agr < M) && (gc < K);
      size_t off = ok ? ((size_t)agr * K + gc) : 0;
      float t = Ab[off];
      ra[j] = ok ? t : 0.f;
    }
  };
  auto loadB = [&](int k0) {
#pragma unroll
    for (int it = 0; it < 4; ++it) {
      int e4   = tid + 256 * it;
      int row  = e4 >> 5;
      int col4 = (e4 & 31) << 2;
      int gk   = k0 + row;
      int gc   = tileN + col4;
      bool ok  = (gk < K) && (gc < Ncols);
      size_t off = ok ? ((size_t)gk * Ncols + gc) : 0;
      fx4 t = *(const fx4*)(Xb + off);
      rb[it] = ok ? t : zero4;
    }
  };
  auto stage = [&](int buf) {
    hx4 pa;
#pragma unroll
    for (int j = 0; j < 4; ++j) pa[j] = (f16)ra[j];
    *(hx4*)&As[buf][arow][acol4] = pa;
#pragma unroll
    for (int it = 0; it < 4; ++it) {
      int e4   = tid + 256 * it;
      int row  = e4 >> 5;
      int col4 = (e4 & 31) << 2;
      *(fx4*)&Bs[buf][row][col4] = rb[it];     // raw f32, ds_store_b128
    }
  };

  v8f acc0 = {}, acc1 = {};
  const int nIter = (K + BK - 1) / BK;
  const int ar  = wm * 16 + mr;
  const int bc0 = wn * 32 + mr;
  const int bc1 = bc0 + 16;

  loadA(0);
  loadB(0);
  stage(0);

  for (int it = 0; it < nIter; ++it) {
    __syncthreads();
    const int cur = it & 1;
    if (it + 1 < nIter) { loadA((it + 1) * BK); loadB((it + 1) * BK); }

    v8u au;
#pragma unroll
    for (int v = 0; v < 8; ++v) {
      int ka = 2 * v + 16 * half;
      au[v] = *(const unsigned*)&As[cur][ar][ka];
    }
    v16h bf0, bf1;
#pragma unroll
    for (int v = 0; v < 8; ++v) {
      int kb = 2 * v + 16 * half;
      bf0[2 * v]     = (f16)Bs[cur][kb][bc0];      // fused into v_cvt_pk_f16_f32
      bf0[2 * v + 1] = (f16)Bs[cur][kb + 1][bc0];
      bf1[2 * v]     = (f16)Bs[cur][kb][bc1];
      bf1[2 * v + 1] = (f16)Bs[cur][kb + 1][bc1];
    }
    v16h af = __builtin_bit_cast(v16h, au);
    acc0 = __builtin_amdgcn_wmma_f32_16x16x32_f16(false, af, false, bf0,
                                                  (short)0, acc0, false, false);
    acc1 = __builtin_amdgcn_wmma_f32_16x16x32_f16(false, af, false, bf1,
                                                  (short)0, acc1, false, false);
    if (it + 1 < nIter) stage(cur ^ 1);
  }

  const int col0 = tileN + bc0;
  const int col1 = tileN + bc1;
#pragma unroll
  for (int v = 0; v < 8; ++v) {
    int r = tileM + wm * 16 + v + 8 * half;
    if (r < M) {
      float s = 1.f, t = convb ? convb[r] : 0.f;
      if (bng) {
        s = bng[r] * rsqrtf(bnv[r] + EPS_BN);
        t = (t - bnm[r]) * s + bnb[r];
      }
      if (col0 < Ncols) Yb[(size_t)r * Ncols + col0] = apply_act(acc0[v] * s + t, act);
      if (col1 < Ncols) Yb[(size_t)r * Ncols + col1] = apply_act(acc1[v] * s + t, act);
    }
  }
}

// ---------------------------------------------------------------------------
// WMMA GEMM with fused graph-feature B operand (B built by KNN gather from
// L2-resident X):  c<Ch : X[c][idx[col]] - X[c][col/Kn]  else X[c-Ch][col/Kn]
// ---------------------------------------------------------------------------
__global__ __launch_bounds__(256) void gemm_gf_bn_act(
    const float* __restrict__ A,
    const float* __restrict__ X, long long strideX,
    const int* __restrict__ idx, long long strideIdx,
    float* __restrict__ Y, long long strideY,
    const float* __restrict__ convb,
    const float* __restrict__ bng, const float* __restrict__ bnb,
    const float* __restrict__ bnm, const float* __restrict__ bnv,
    int M, int Ch, int Np, int Kn, int Ncols, int act)
{
  __shared__ f16   As[2][BM][BK + APAD];
  __shared__ float Bs[2][BK][BN + BPADF];
  __shared__ int   ctrN[BN];
  __shared__ int   nbrN[BN];

  const int b     = blockIdx.z;
  const int tileN = blockIdx.x * BN;
  const int tileM = blockIdx.y * BM;
  const int tid   = threadIdx.x;
  const int lane  = tid & 31;
  const int wave  = tid >> 5;
  const int wm    = wave & 1;
  const int wn    = wave >> 1;
  const int half  = lane >> 4;
  const int mr    = lane & 15;
  const int Ktot  = 2 * Ch;            // multiple of 32

  const float* Xb   = X + (size_t)b * strideX;
  const int*   idxb = idx + (size_t)b * strideIdx;
  float*       Yb   = Y + (size_t)b * strideY;

  if (tid < BN) {
    int gcol = tileN + tid;
    int n = 0, nb = 0;
    if (gcol < Ncols) { n = gcol / Kn; nb = idxb[gcol]; }
    ctrN[tid] = n;
    nbrN[tid] = nb;
  }
  __syncthreads();

  const int arow  = tid >> 3;
  const int acol4 = (tid & 7) << 2;
  const int agr   = tileM + arow;

  float ra[4];
  float rc[16], rn[16];

  auto loadA = [&](int k0) {
#pragma unroll
    for (int j = 0; j < 4; ++j) {
      int gc = k0 + acol4 + j;
      bool ok = (agr < M) && (gc < Ktot);
      size_t off = ok ? ((size_t)agr * Ktot + gc) : 0;
      float t = A[off];
      ra[j] = ok ? t : 0.f;
    }
  };

  auto loadB = [&](int k0) {
    const bool lower = (k0 < Ch);      // uniform: Ch % 32 == 0
#pragma unroll
    for (int it = 0; it < 4; ++it) {
#pragma unroll
      for (int j = 0; j < 4; ++j) {
        int e4  = tid + 256 * it;
        int row = e4 >> 5;
        int col = ((e4 & 31) << 2) + j;
        int gk  = k0 + row;
        int gcol = tileN + col;
        bool ok = (gcol < Ncols);
        int n  = ctrN[col];
        int nb = nbrN[col];
        int q  = it * 4 + j;
        if (lower) {
          float tc = Xb[(size_t)gk * Np + n];
          float tn = Xb[(size_t)gk * Np + nb];
          rc[q] = ok ? tc : 0.f;
          rn[q] = ok ? tn : 0.f;
        } else {
          float tc = Xb[(size_t)(gk - Ch) * Np + n];
          rc[q] = ok ? tc : 0.f;
          rn[q] = 0.f;
        }
      }
    }
  };
  auto stage = [&](int buf, int k0) {
    const bool lower = (k0 < Ch);
    hx4 pa;
#pragma unroll
    for (int j = 0; j < 4; ++j) pa[j] = (f16)ra[j];
    *(hx4*)&As[buf][arow][acol4] = pa;
#pragma unroll
    for (int it = 0; it < 4; ++it) {
      int e4   = tid + 256 * it;
      int row  = e4 >> 5;
      int col4 = (e4 & 31) << 2;
      fx4 pb;
#pragma unroll
      for (int j = 0; j < 4; ++j) {
        int q = it * 4 + j;
        pb[j] = lower ? (rn[q] - rc[q]) : rc[q];
      }
      *(fx4*)&Bs[buf][row][col4] = pb;           // raw f32, ds_store_b128
    }
  };

  v8f acc0 = {}, acc1 = {};
  const int nIter = Ktot / BK;
  const int ar  = wm * 16 + mr;
  const int bc0 = wn * 32 + mr;
  const int bc1 = bc0 + 16;

  loadA(0);
  loadB(0);
  stage(0, 0);

  for (int it = 0; it < nIter; ++it) {
    __syncthreads();
    const int cur = it & 1;
    if (it + 1 < nIter) { loadA((it + 1) * BK); loadB((it + 1) * BK); }

    v8u au;
#pragma unroll
    for (int v = 0; v < 8; ++v) {
      int ka = 2 * v + 16 * half;
      au[v] = *(const unsigned*)&As[cur][ar][ka];
    }
    v16h bf0, bf1;
#pragma unroll
    for (int v = 0; v < 8; ++v) {
      int kb = 2 * v + 16 * half;
      bf0[2 * v]     = (f16)Bs[cur][kb][bc0];
      bf0[2 * v + 1] = (f16)Bs[cur][kb + 1][bc0];
      bf1[2 * v]     = (f16)Bs[cur][kb][bc1];
      bf1[2 * v + 1] = (f16)Bs[cur][kb + 1][bc1];
    }
    v16h af = __builtin_bit_cast(v16h, au);
    acc0 = __builtin_amdgcn_wmma_f32_16x16x32_f16(false, af, false, bf0,
                                                  (short)0, acc0, false, false);
    acc1 = __builtin_amdgcn_wmma_f32_16x16x32_f16(false, af, false, bf1,
                                                  (short)0, acc1, false, false);
    if (it + 1 < nIter) stage(cur ^ 1, (it + 1) * BK);
  }

  const int col0 = tileN + bc0;
  const int col1 = tileN + bc1;
#pragma unroll
  for (int v = 0; v < 8; ++v) {
    int r = tileM + wm * 16 + v + 8 * half;
    if (r < M) {
      float s = 1.f, t = convb ? convb[r] : 0.f;
      if (bng) {
        s = bng[r] * rsqrtf(bnv[r] + EPS_BN);
        t = (t - bnm[r]) * s + bnb[r];
      }
      if (col0 < Ncols) Yb[(size_t)r * Ncols + col0] = apply_act(acc0[v] * s + t, act);
      if (col1 < Ncols) Yb[(size_t)r * Ncols + col1] = apply_act(acc1[v] * s + t, act);
    }
  }
}

// max over trailing K neighbors: in [C][Np*Kn] -> out [C][Np] (per batch)
__global__ void kmax_kernel(const float* __restrict__ in, long long inB,
                            float* __restrict__ out, long long outB,
                            int C, int Np, int Kn)
{
  int i = blockIdx.x * blockDim.x + threadIdx.x;
  if (i >= C * Np) return;
  int b = blockIdx.z;
  int c = i / Np, n = i % Np;
  const float* p = in + (size_t)b * inB + (size_t)c * Np * Kn + (size_t)n * Kn;
  float m = p[0];
  for (int k = 1; k < Kn; ++k) m = fmaxf(m, p[k]);
  out[(size_t)b * outB + (size_t)c * Np + n] = m;
}

// max over N: in [C][Np] (batch stride inB) -> out [b*C + c]
__global__ void rowmax_kernel(const float* __restrict__ in, long long inB,
                              float* __restrict__ out, int C, int Np)
{
  __shared__ float red[256];
  int b = blockIdx.z, c = blockIdx.x;
  const float* p = in + (size_t)b * inB + (size_t)c * Np;
  float m = -INFINITY;
  for (int n = threadIdx.x; n < Np; n += 256) m = fmaxf(m, p[n]);
  red[threadIdx.x] = m;
  __syncthreads();
  for (int s = 128; s > 0; s >>= 1) {
    if (threadIdx.x < s) red[threadIdx.x] = fmaxf(red[threadIdx.x], red[threadIdx.x + s]);
    __syncthreads();
  }
  if (threadIdx.x == 0) out[b * C + c] = red[0];
}

// tiny FC (batch=2)
__global__ void fc_bn_act(const float* __restrict__ in, const float* __restrict__ W,
                          const float* __restrict__ convb,
                          const float* __restrict__ g, const float* __restrict__ bb,
                          const float* __restrict__ m, const float* __restrict__ v,
                          float* __restrict__ out, int K, int O, int act, int addEye)
{
  int i = blockIdx.x * blockDim.x + threadIdx.x;
  if (i >= 2 * O) return;
  int b = i / O, o = i % O;
  const float* x = in + (size_t)b * K;
  float acc = convb ? convb[o] : 0.f;
  for (int k = 0; k < K; ++k) acc += W[(size_t)o * K + k] * x[k];
  if (g) { float s = g[o] * rsqrtf(v[o] + EPS_BN); acc = (acc - m[o]) * s + bb[o]; }
  if (addEye && (o % 65 == 0)) acc += 1.f;
  out[i] = apply_act(acc, act);
}

// transT[b][d][c] = trans[b][c][d]
__global__ void transpose64(const float* __restrict__ in, float* __restrict__ out)
{
  int i = blockIdx.x * blockDim.x + threadIdx.x;
  if (i >= 2 * 4096) return;
  int b = i >> 12, r = (i >> 6) & 63, c = i & 63;
  out[i] = in[b * 4096 + c * 64 + r];
}

// cat rows 0..511 = broadcast gmax
__global__ void broadcast_rows(const float* __restrict__ gmax,
                               float* __restrict__ cat, int Np)
{
  long long i = (long long)blockIdx.x * blockDim.x + threadIdx.x;
  if (i >= 2LL * 512 * Np) return;
  int b = (int)(i / (512LL * Np));
  long long rem = i % (512LL * Np);
  int c = (int)(rem / Np), n = (int)(rem % Np);
  cat[(size_t)b * 1600 * Np + (size_t)c * Np + n] = gmax[b * 512 + c];
}

// softmax over 15 classes + transpose to [b][n][15]
__global__ void softmax15(const float* __restrict__ pre, float* __restrict__ out, int Np)
{
  int i = blockIdx.x * blockDim.x + threadIdx.x;
  if (i >= 2 * Np) return;
  int b = i / Np, n = i % Np;
  const float* p = pre + (size_t)b * 15 * Np + n;
  float vals[15], mx = -INFINITY;
#pragma unroll
  for (int o = 0; o < 15; ++o) { vals[o] = p[(size_t)o * Np]; mx = fmaxf(mx, vals[o]); }
  float s = 0.f;
#pragma unroll
  for (int o = 0; o < 15; ++o) { vals[o] = __expf(vals[o] - mx); s += vals[o]; }
  float inv = 1.f / s;
  float* q = out + ((size_t)b * Np + n) * 15;
#pragma unroll
  for (int o = 0; o < 15; ++o) q[o] = vals[o] * inv;
}

// ---------------------------------------------------------------------------
extern "C" void kernel_launch(void* const* d_in, const int* in_sizes, int n_in,
                              void* d_out, int out_size, void* d_ws, size_t ws_size,
                              hipStream_t stream)
{
  const int Np = 10000;
  const long long CATB = 1600LL * Np;

  auto P = [&](int i) { return (const float*)d_in[i]; };
  const float* x0    = (const float*)d_in[0];
  const int*   knn12 = (const int*)d_in[1];
  const int*   knn6  = (const int*)d_in[2];

  char* ws = (char*)d_ws;
  const size_t OFF_CAT   = 0;
  const size_t OFF_H1    = OFF_CAT  + 2ull * 1600 * Np * 4;
  const size_t OFF_H2    = OFF_H1   + 2ull * 64 * Np * 4;
  const size_t OFF_BUFA  = OFF_H2   + 2ull * 64 * Np * 4;
  const size_t OFF_BUFB  = OFF_BUFA + 2ull * 512 * Np * 4;
  const size_t OFF_SMALL = OFF_BUFB + 2ull * 512 * Np * 4;
  const size_t OFF_BIG0  = OFF_SMALL + (1ull << 20);
  const size_t OFF_BIG1  = OFF_BIG0 + 2ull * 512 * 120000 * 4;

  float* cat    = (float*)(ws + OFF_CAT);
  float* h1     = (float*)(ws + OFF_H1);
  float* h2     = (float*)(ws + OFF_H2);
  float* bufA   = (float*)(ws + OFF_BUFA);
  float* bufB   = (float*)(ws + OFF_BUFB);
  float* pooled = (float*)(ws + OFF_SMALL);
  float* fcb1   = (float*)(ws + OFF_SMALL + 8192);
  float* fcb2   = (float*)(ws + OFF_SMALL + 16384);
  float* trans  = (float*)(ws + OFF_SMALL + 32768);
  float* transT = (float*)(ws + OFF_SMALL + 65536);
  float* gmax   = (float*)(ws + OFF_SMALL + 98304);
  float* big0   = (float*)(ws + OFF_BIG0);
  float* big1   = (float*)(ws + OFF_BIG1);

  auto gemm = [&](const float* A, long long sA, const float* X, long long sX,
                  float* Y, long long sY, const float* cb, const float* g,
                  const float* bb, const float* m, const float* v,
                  int M, int K, int Nc, int act) {
    dim3 grid((Nc + BN - 1) / BN, (M + BM - 1) / BM, 2);
    gemm_bn_act<<<grid, 256, 0, stream>>>(A, sA, X, sX, Y, sY, cb, g, bb, m, v,
                                          M, K, Nc, act);
  };
  auto gemm_gf = [&](const float* A, const float* X, long long sX,
                     const int* idx, long long sIdx, float* Y, long long sY,
                     const float* cb, const float* g, const float* bb,
                     const float* m, const float* v,
                     int M, int Ch, int Kn, int Nc, int act) {
    dim3 grid((Nc + BN - 1) / BN, (M + BM - 1) / BM, 2);
    gemm_gf_bn_act<<<grid, 256, 0, stream>>>(A, X, sX, idx, sIdx, Y, sY, cb,
                                             g, bb, m, v, M, Ch, Np, Kn, Nc, act);
  };
  auto kmax = [&](const float* in, long long inB, float* out, long long outB,
                  int C, int Kn) {
    dim3 grid((C * Np + 255) / 256, 1, 2);
    kmax_kernel<<<grid, 256, 0, stream>>>(in, inB, out, outB, C, Np, Kn);
  };

  // ---- mlp1 ----
  gemm(P(3), 0, x0, 15LL * Np, h1, 64LL * Np, P(4), P(5), P(6), P(7), P(8), 64, 15, Np, 1);
  gemm(P(9), 0, h1, 64LL * Np, h2, 64LL * Np, P(10), P(11), P(12), P(13), P(14), 64, 64, Np, 1);

  // ---- fstn ----
  gemm(P(15), 0, h2, 64LL * Np, bufA, 64LL * Np, P(16), P(17), P(18), P(19), P(20), 64, 64, Np, 1);
  gemm(P(21), 0, bufA, 64LL * Np, bufB, 128LL * Np, P(22), P(23), P(24), P(25), P(26), 128, 64, Np, 1);
  gemm(P(27), 0, bufB, 128LL * Np, bufA, 256LL * Np, P(28), P(29), P(30), P(31), P(32), 256, 128, Np, 1);
  gemm(P(33), 0, bufA, 256LL * Np, bufB, 512LL * Np, P(34), P(35), P(36), P(37), P(38), 512, 256, Np, 1);
  rowmax_kernel<<<dim3(512, 1, 2), 256, 0, stream>>>(bufB, 512LL * Np, pooled, 512, Np);
  fc_bn_act<<<2, 256, 0, stream>>>(pooled, P(39), P(40), P(41), P(42), P(43), P(44), fcb1, 512, 256, 1, 0);
  fc_bn_act<<<1, 256, 0, stream>>>(fcb1, P(45), P(46), P(47), P(48), P(49), P(50), fcb2, 256, 128, 1, 0);
  fc_bn_act<<<32, 256, 0, stream>>>(fcb2, P(51), P(52), nullptr, nullptr, nullptr, nullptr, trans, 128, 4096, 0, 1);
  transpose64<<<32, 256, 0, stream>>>(trans, transT);

  // ---- x_ftm -> cat rows 512..575 ----
  gemm(transT, 4096, h2, 64LL * Np, cat + 512LL * Np, CATB,
       nullptr, nullptr, nullptr, nullptr, nullptr, 64, 64, Np, 0);

  // ---- ec1 ----
  gemm_gf(P(53), cat + 512LL * Np, CATB, knn12, 12LL * Np, big0, 64LL * 120000,
          P(54), P(55), P(56), P(57), P(58), 64, 64, 12, 120000, 2);
  gemm(P(59), 0, big0, 64LL * 120000, big1, 64LL * 120000,
       P(60), P(61), P(62), P(63), P(64), 64, 64, 120000, 2);
  kmax(big1, 64LL * 120000, h1, 64LL * Np, 64, 12);

  // ---- mlp2 ----
  gemm(P(65), 0, h1, 64LL * Np, h2, 64LL * Np, P(66), P(67), P(68), P(69), P(70), 64, 64, Np, 1);
  gemm(P(71), 0, h2, 64LL * Np, bufA, 128LL * Np, P(72), P(73), P(74), P(75), P(76), 128, 64, Np, 1);
  gemm(P(77), 0, bufA, 128LL * Np, bufB, 256LL * Np, P(78), P(79), P(80), P(81), P(82), 256, 128, Np, 1);
  gemm(P(83), 0, bufB, 256LL * Np, cat + 576LL * Np, CATB,
       P(84), P(85), P(86), P(87), P(88), 512, 256, Np, 1);

  // ---- ec2 branch 1 (knn12) ----
  gemm_gf(P(89), cat + 576LL * Np, CATB, knn12, 12LL * Np, big0, 512LL * 120000,
          P(90), P(91), P(92), P(93), P(94), 512, 512, 12, 120000, 2);
  gemm(P(95), 0, big0, 512LL * 120000, big1, 128LL * 120000,
       P(96), P(97), P(98), P(99), P(100), 128, 512, 120000, 2);
  kmax(big1, 128LL * 120000, bufA, 256LL * Np, 128, 12);

  // ---- ec2 branch 2 (knn6) ----
  gemm_gf(P(101), cat + 576LL * Np, CATB, knn6, 6LL * Np, big0, 512LL * 60000,
          P(102), P(103), P(104), P(105), P(106), 512, 512, 6, 60000, 2);
  gemm(P(107), 0, big0, 512LL * 60000, big1, 128LL * 60000,
       P(108), P(109), P(110), P(111), P(112), 128, 512, 60000, 2);
  kmax(big1, 128LL * 60000, bufA + 128LL * Np, 256LL * Np, 128, 6);

  // ---- x_glm2 -> cat rows 1088..1599 ----
  gemm(P(113), 0, bufA, 256LL * Np, cat + 1088LL * Np, CATB,
       P(114), P(115), P(116), P(117), P(118), 512, 256, Np, 1);

  // ---- global max + broadcast ----
  rowmax_kernel<<<dim3(512, 1, 2), 256, 0, stream>>>(cat + 1088LL * Np, CATB, gmax, 512, Np);
  broadcast_rows<<<(int)((2LL * 512 * Np + 255) / 256), 256, 0, stream>>>(gmax, cat, Np);

  // ---- mlp3 + out + softmax ----
  gemm(P(119), 0, cat, CATB, big0, 256LL * Np, P(120), P(121), P(122), P(123), P(124), 256, 1600, Np, 1);
  gemm(P(125), 0, big0, 256LL * Np, big1, 256LL * Np, P(126), P(127), P(128), P(129), P(130), 256, 256, Np, 1);
  gemm(P(131), 0, big1, 256LL * Np, big0, 128LL * Np, P(132), P(133), P(134), P(135), P(136), 128, 256, Np, 1);
  gemm(P(137), 0, big0, 128LL * Np, big1, 128LL * Np, P(138), P(139), P(140), P(141), P(142), 128, 128, Np, 1);
  gemm(P(143), 0, big1, 128LL * Np, big0, 15LL * Np,
       P(144), nullptr, nullptr, nullptr, nullptr, 15, 128, Np, 0);
  softmax15<<<(2 * Np + 255) / 256, 256, 0, stream>>>(big0, (float*)d_out, Np);
}
